// speechsplit_89146341195964
// MI455X (gfx1250) — matrix operand order repaired and checked by reading, not compiled, not run on hardware
//
#include <hip/hip_runtime.h>
#include <hip/hip_bf16.h>
#include <math.h>

// ---------------------------------------------------------------------------
// CDNA5 (gfx1250) wave32 WMMA types
// ---------------------------------------------------------------------------
typedef __attribute__((ext_vector_type(16))) _Float16 v16h;
typedef __attribute__((ext_vector_type(8)))  float    v8f;

#define WAVE 32

// ---------------------------------------------------------------------------
// WMMA fragment loaders (contiguous k order shared by A and B, wave32)
// A fragment: 16x32 f16, row m = lane%16; lanes0-15 hold K=0..15,
// lanes16-31 hold K=16..31 (16 consecutive halves per lane).
// ---------------------------------------------------------------------------
__device__ __forceinline__ v16h load_a_frag(const _Float16* A, int row0, int ld, int k0) {
    int lane = threadIdx.x & (WAVE - 1);
    const _Float16* p = A + (size_t)(row0 + (lane & 15)) * ld;
    int h16 = (lane >> 4) * 16;
    v16h a;
#pragma unroll
    for (int e = 0; e < 16; ++e) a[e] = p[k0 + h16 + e];
    return a;
}

// B fragment: 32x16 f16, col n = lane%16; lanes0-15 hold K=0..15,
// lanes16-31 hold K=16..31 (16 consecutive halves per lane).
// Bt is stored transposed: [N][Kpad] row-major -> contiguous 32B per lane.
__device__ __forceinline__ v16h load_b_frag(const _Float16* Bt, int n0, int ld, int k0) {
    int lane = threadIdx.x & (WAVE - 1);
    const _Float16* p = Bt + (size_t)(n0 + (lane & 15)) * ld + k0 + ((lane >> 4) << 4);
    v16h b;
#pragma unroll
    for (int e = 0; e < 16; ++e) b[e] = p[e];
    return b;
}

__device__ __forceinline__ float sigf(float x) { return 1.0f / (1.0f + __expf(-x)); }

// ---------------------------------------------------------------------------
// Generic WMMA GEMM: C[M,N] = A[M,K] (f16, lda) * Bt[N,K]^T (f16, ldb)
// + optional bias (+ optional ReLU). One wave per 16x16 tile.
// M, N multiples of 16; K multiple of 32.
// ---------------------------------------------------------------------------
__global__ __launch_bounds__(256) void wmma_gemm_kernel(
    const _Float16* __restrict__ A, int lda,
    const _Float16* __restrict__ Bt, int ldb,
    float* __restrict__ C, int ldc,
    int N, int K,
    const float* __restrict__ bias, int biasN, int relu)
{
    int wave = threadIdx.x >> 5;
    int lane = threadIdx.x & 31;
    int m0 = blockIdx.x * 16;
    int n0 = (blockIdx.y * (blockDim.x >> 5) + wave) * 16;
    if (n0 >= N) return;

    v8f acc = {};
    for (int k0 = 0; k0 < K; k0 += 32) {
        v16h a = load_a_frag(A, m0, lda, k0);
        v16h b = load_b_frag(Bt, n0, ldb, k0);
        acc = __builtin_amdgcn_wmma_f32_16x16x32_f16(false, a, false, b,
                                                     (short)0, acc, false, false);
    }
    int ncol = n0 + (lane & 15);
    float bv = 0.0f;
    if (bias && ncol < biasN) bv = bias[ncol];
#pragma unroll
    for (int j = 0; j < 8; ++j) {
        int m = m0 + j + ((lane >> 4) << 3);
        float v = acc[j] + bv;
        if (relu) v = fmaxf(v, 0.0f);
        C[(size_t)m * ldc + ncol] = v;
    }
}

// ---------------------------------------------------------------------------
// Persistent decoder LSTM (hidden 512, batch 16). grid = {2} (fwd/bwd),
// block = 1024 (32 waves). h lives in LDS as the WMMA A operand (16x512 f16);
// the cell state c stays in D-fragment-layout registers. Each wave owns 16
// hidden units and computes the 4 gate tiles with v_wmma_f32_16x16x32_f16.
// The next timestep's xs rows are prefetched (global_prefetch_b8) before the
// gate GEMM chain so their latency hides under the WMMA work.
// ---------------------------------------------------------------------------
#define DH 512
__global__ __launch_bounds__(1024) void decoder_lstm_kernel(
    const float* __restrict__ xsF, const float* __restrict__ xsB,   // (B,T,2048) f32, bias included
    const _Float16* __restrict__ whhF, const _Float16* __restrict__ whhB, // [2048][512] f16
    _Float16* __restrict__ outA,  // (B*T, 1024) f16, fwd cols 0:512, bwd 512:1024
    int T)
{
    int dir = blockIdx.x;
    const float* xs = dir ? xsB : xsF;
    const _Float16* whh = dir ? whhB : whhF;

    __shared__ _Float16 h_lds[16 * DH];
    for (int i = threadIdx.x; i < 16 * DH; i += blockDim.x) h_lds[i] = (_Float16)0.0f;
    __syncthreads();

    int wave = threadIdx.x >> 5;
    int lane = threadIdx.x & 31;
    int j0 = wave * 16;             // this wave's hidden-unit block
    int u  = j0 + (lane & 15);      // hidden unit for this lane column
    int bq = (lane >> 4) << 3;      // batch row offset for D-fragment

    v8f cst = {};                   // cell state, D-fragment layout (b = j + bq)

    for (int s = 0; s < T; ++s) {
        int t = dir ? (T - 1 - s) : s;

        // prefetch next step's xs (4 gate segments per batch row we will read)
        if (s + 1 < T) {
            int tn = dir ? (T - 2 - s) : (s + 1);
#pragma unroll
            for (int j = 0; j < 8; ++j) {
                int b = j + bq;
                const float* p = xs + ((size_t)(b * T + tn)) * 2048 + u;
                __builtin_prefetch(p + 0 * DH, 0, 1);
                __builtin_prefetch(p + 1 * DH, 0, 1);
                __builtin_prefetch(p + 2 * DH, 0, 1);
                __builtin_prefetch(p + 3 * DH, 0, 1);
            }
        }

        v8f g0 = {}, g1 = {}, g2 = {}, g3 = {};
#pragma unroll 2
        for (int kb = 0; kb < DH; kb += 32) {
            v16h a = load_a_frag(h_lds, 0, DH, kb);   // A = h (16 x 512), from LDS
            v16h b0 = load_b_frag(whh, 0 * DH + j0, DH, kb);
            g0 = __builtin_amdgcn_wmma_f32_16x16x32_f16(false, a, false, b0, (short)0, g0, false, false);
            v16h b1 = load_b_frag(whh, 1 * DH + j0, DH, kb);
            g1 = __builtin_amdgcn_wmma_f32_16x16x32_f16(false, a, false, b1, (short)0, g1, false, false);
            v16h b2 = load_b_frag(whh, 2 * DH + j0, DH, kb);
            g2 = __builtin_amdgcn_wmma_f32_16x16x32_f16(false, a, false, b2, (short)0, g2, false, false);
            v16h b3 = load_b_frag(whh, 3 * DH + j0, DH, kb);
            g3 = __builtin_amdgcn_wmma_f32_16x16x32_f16(false, a, false, b3, (short)0, g3, false, false);
        }
        __syncthreads();   // everyone done reading h_lds

#pragma unroll
        for (int j = 0; j < 8; ++j) {
            int b = j + bq;
            size_t base = ((size_t)(b * T + t)) * 2048;
            float gi = g0[j] + xs[base + 0 * DH + u];
            float gf = g1[j] + xs[base + 1 * DH + u];
            float gg = g2[j] + xs[base + 2 * DH + u];
            float go = g3[j] + xs[base + 3 * DH + u];
            float cv = sigf(gf) * cst[j] + sigf(gi) * tanhf(gg);
            cst[j] = cv;
            float hv = sigf(go) * tanhf(cv);
            _Float16 hh = (_Float16)hv;
            h_lds[b * DH + u] = hh;
            outA[((size_t)(b * T + t)) * 1024 + dir * DH + u] = hh;
        }
        __syncthreads();   // h_lds updated for next step
    }
}

// ---------------------------------------------------------------------------
// Small bi-LSTM (H <= 32): grid {B, 2}, block 128. xs already has bias.
// ---------------------------------------------------------------------------
__global__ void small_lstm_kernel(
    const float* __restrict__ xsF, const float* __restrict__ xsB, int ldxs,
    const float* __restrict__ whhF, const float* __restrict__ whhB,  // [4H][H] f32
    float* __restrict__ out, int ldo, int H, int T)
{
    int b = blockIdx.x, dir = blockIdx.y;
    const float* xs  = dir ? xsB : xsF;
    const float* whh = dir ? whhB : whhF;
    __shared__ float hsh[32];
    __shared__ float gsh[128];
    int n = threadIdx.x;
    float cst = 0.0f;
    if (n < H) hsh[n] = 0.0f;
    __syncthreads();
    for (int s = 0; s < T; ++s) {
        int t = dir ? (T - 1 - s) : s;
        if (n < 4 * H) {
            float gv = xs[(size_t)(b * T + t) * ldxs + n];
            for (int u = 0; u < H; ++u) gv += whh[n * H + u] * hsh[u];
            gsh[n] = gv;
        }
        __syncthreads();
        if (n < H) {
            float iv = sigf(gsh[n]);
            float fv = sigf(gsh[H + n]);
            float gg = tanhf(gsh[2 * H + n]);
            float ov = sigf(gsh[3 * H + n]);
            cst = fv * cst + iv * gg;
            float hv = ov * tanhf(cst);
            hsh[n] = hv;
            out[(size_t)(b * T + t) * ldo + dir * H + n] = hv;
        }
        __syncthreads();
    }
}

// ---------------------------------------------------------------------------
// im2col for conv1d k=5 pad=2 over time, fp32 -> fp16, K = kk*Cin + cin,
// zero-padded to Kpad.
// ---------------------------------------------------------------------------
__global__ void im2col5_kernel(const float* __restrict__ src, int lds, int coff,
                               int Cin, int Kpad, _Float16* __restrict__ dst,
                               int T, long total)
{
    long i = (long)blockIdx.x * blockDim.x + threadIdx.x;
    long stride = (long)gridDim.x * blockDim.x;
    int KC = 5 * Cin;
    for (; i < total; i += stride) {
        int r = (int)(i / Kpad);
        int k = (int)(i % Kpad);
        float v = 0.0f;
        if (k < KC) {
            int kk = k / Cin, cin = k % Cin;
            int t = r % T, b = r / T;
            int ts = t + kk - 2;
            if (ts >= 0 && ts < T) v = src[(size_t)(b * T + ts) * lds + coff + cin];
        }
        dst[i] = (_Float16)v;
    }
}

// conv weight (O,Cin,5) -> Bt [O][Kpad] with K = kk*Cin+cin, fp16
__global__ void cvt_convw_kernel(const float* __restrict__ w, int Cin, int Kpad,
                                 _Float16* __restrict__ dst, long total)
{
    long i = (long)blockIdx.x * blockDim.x + threadIdx.x;
    long stride = (long)gridDim.x * blockDim.x;
    int KC = 5 * Cin;
    for (; i < total; i += stride) {
        int o = (int)(i / Kpad);
        int k = (int)(i % Kpad);
        float v = 0.0f;
        if (k < KC) {
            int kk = k / Cin, cin = k % Cin;
            v = w[((size_t)o * Cin + cin) * 5 + kk];
        }
        dst[i] = (_Float16)v;
    }
}

// dense weight (N,K) f32 -> Bt [Npad][Kpad] f16 (zero-padded)
__global__ void cvt_w_kernel(const float* __restrict__ w, int N, int K, int Kpad,
                             _Float16* __restrict__ dst, long total)
{
    long i = (long)blockIdx.x * blockDim.x + threadIdx.x;
    long stride = (long)gridDim.x * blockDim.x;
    for (; i < total; i += stride) {
        int n = (int)(i / Kpad);
        int k = (int)(i % Kpad);
        float v = (n < N && k < K) ? w[(size_t)n * K + k] : 0.0f;
        dst[i] = (_Float16)v;
    }
}

// activation slice (M rows, ld, coff, width K) f32 -> f16 [M][Kpad]
__global__ void cvt_pad_kernel(const float* __restrict__ src, int lds, int coff,
                               int K, int Kpad, _Float16* __restrict__ dst, long total)
{
    long i = (long)blockIdx.x * blockDim.x + threadIdx.x;
    long stride = (long)gridDim.x * blockDim.x;
    for (; i < total; i += stride) {
        int r = (int)(i / Kpad);
        int k = (int)(i % Kpad);
        dst[i] = (_Float16)((k < K) ? src[(size_t)r * lds + coff + k] : 0.0f);
    }
}

// GroupNorm (+ReLU) in place over (t, c-in-group); grid {B, G}, block 256
__global__ void groupnorm_relu_kernel(float* __restrict__ x, int ld, int coff, int C,
                                      int T, const float* __restrict__ gamma,
                                      const float* __restrict__ beta)
{
    int b = blockIdx.x, g = blockIdx.y, G = gridDim.y;
    int Cg = C / G;
    int n = Cg * T;
    __shared__ float s1[256], s2[256];
    float a0 = 0.0f, a1 = 0.0f;
    for (int i = threadIdx.x; i < n; i += 256) {
        int t = i / Cg, c = i % Cg;
        float v = x[(size_t)(b * T + t) * ld + coff + g * Cg + c];
        a0 += v; a1 += v * v;
    }
    s1[threadIdx.x] = a0; s2[threadIdx.x] = a1;
    __syncthreads();
    for (int s = 128; s > 0; s >>= 1) {
        if (threadIdx.x < s) { s1[threadIdx.x] += s1[threadIdx.x + s]; s2[threadIdx.x] += s2[threadIdx.x + s]; }
        __syncthreads();
    }
    float mean = s1[0] / n;
    float var  = s2[0] / n - mean * mean;
    float inv  = rsqrtf(var + 1e-5f);
    for (int i = threadIdx.x; i < n; i += 256) {
        int t = i / Cg, c = i % Cg;
        size_t idx = (size_t)(b * T + t) * ld + coff + g * Cg + c;
        int ch = g * Cg + c;
        float v = (x[idx] - mean) * inv * gamma[ch] + beta[ch];
        x[idx] = fmaxf(v, 0.0f);
    }
}

// Build decoder input: [codes_c(16) | codes_r(2) | codes_f(64) | c_trg(82)] = 164
__global__ void build_enc_kernel(const float* __restrict__ co,  // (BT,16)
                                 const float* __restrict__ ro,  // (BT,2)
                                 const float* __restrict__ fo,  // (BT,64)
                                 const float* __restrict__ ctrg,// (B,82)
                                 float* __restrict__ enc, int T, long total)
{
    long i = (long)blockIdx.x * blockDim.x + threadIdx.x;
    long stride = (long)gridDim.x * blockDim.x;
    for (; i < total; i += stride) {
        int r = (int)(i / 164), ch = (int)(i % 164);
        int b = r / T, t = r % T, t8 = t / 8;
        int rl = b * T + t8 * 8 + 7;   // codes use last frame of octet ...
        int rf = b * T + t8 * 8;       // ... or first frame
        float v;
        if      (ch < 8)   v = co[(size_t)rl * 16 + ch];
        else if (ch < 16)  v = co[(size_t)rf * 16 + ch];
        else if (ch == 16) v = ro[(size_t)rl * 2 + 0];
        else if (ch == 17) v = ro[(size_t)rf * 2 + 1];
        else if (ch < 50)  v = fo[(size_t)rl * 64 + (ch - 18)];
        else if (ch < 82)  v = fo[(size_t)rf * 64 + (ch - 18)];
        else               v = ctrg[(size_t)b * 82 + (ch - 82)];
        enc[i] = v;
    }
}

// ---------------------------------------------------------------------------
// Host-side launch helpers
// ---------------------------------------------------------------------------
static inline int pad32(int k) { return (k + 31) & ~31; }

static inline void launch_gemm(const _Float16* A, int lda, const _Float16* Bt, int ldb,
                               float* C, int ldc, int M, int N, int K,
                               const float* bias, int biasN, int relu, hipStream_t s)
{
    dim3 g(M / 16, (N / 16 + 7) / 8);
    wmma_gemm_kernel<<<g, 256, 0, s>>>(A, lda, Bt, ldb, C, ldc, N, K, bias, biasN, relu);
}

static inline dim3 ew_grid(long total) {
    long b = (total + 255) / 256;
    if (b > 32768) b = 32768;
    return dim3((unsigned)b);
}

struct Bump { char* p; size_t off; };
static inline void* balloc(Bump& w, size_t bytes) {
    size_t o = (w.off + 255) & ~(size_t)255;
    w.off = o + bytes;
    return (void*)(w.p + o);
}

extern "C" void kernel_launch(void* const* d_in, const int* in_sizes, int n_in,
                              void* d_out, int out_size, void* d_ws, size_t ws_size,
                              hipStream_t stream)
{
    (void)in_sizes; (void)n_in; (void)out_size; (void)ws_size;
    const int B = 16, T = 192, BT = B * T;

    auto IN = [&](int i) { return (const float*)d_in[i]; };
    // input leaf indices (setup_inputs dict order, pytree-flattened)
    // 0 x_f0, 1 x_org, 2 c_trg, 3-5 conv_c_w, 6-8 conv_c_b, 9-10 gn_c,
    // 11-13 conv_f_w, 14-16 conv_f_b, 17-18 gn_f, 19 conv_r_w, 20 conv_r_b,
    // 21-22 gn_r, 23-28 lstm_c, 29-34 lstm_f, 35-40 lstm_r,
    // 41-58 lstm_d (3 layers x (wf,uf,bf,wb,ub,bb)), 59 lin_w, 60 lin_b

    Bump ws{(char*)d_ws, 0};
    _Float16* Aim   = (_Float16*)balloc(ws, (size_t)BT * 2560 * sizeof(_Float16)); // im2col / act f16
    float*    cfA   = (float*)   balloc(ws, (size_t)BT * 768 * sizeof(float));
    float*    cfB   = (float*)   balloc(ws, (size_t)BT * 768 * sizeof(float));
    _Float16* Wbuf  = (_Float16*)balloc(ws, (size_t)2048 * 1024 * sizeof(_Float16));
    _Float16* whhF  = (_Float16*)balloc(ws, (size_t)2048 * 512 * sizeof(_Float16));
    _Float16* whhB  = (_Float16*)balloc(ws, (size_t)2048 * 512 * sizeof(_Float16));
    float*    xsF   = (float*)   balloc(ws, (size_t)BT * 2048 * sizeof(float));
    float*    xsB   = (float*)   balloc(ws, (size_t)BT * 2048 * sizeof(float));
    _Float16* actE0 = (_Float16*)balloc(ws, (size_t)BT * 1024 * sizeof(_Float16));
    _Float16* actE1 = (_Float16*)balloc(ws, (size_t)BT * 1024 * sizeof(_Float16));
    _Float16* encH  = (_Float16*)balloc(ws, (size_t)BT * 192 * sizeof(_Float16));
    float*    co    = (float*)   balloc(ws, (size_t)BT * 16 * sizeof(float));
    float*    fo    = (float*)   balloc(ws, (size_t)BT * 64 * sizeof(float));
    float*    ro    = (float*)   balloc(ws, (size_t)BT * 2 * sizeof(float));
    float*    rbuf  = (float*)   balloc(ws, (size_t)BT * 128 * sizeof(float));
    float*    encF  = (float*)   balloc(ws, (size_t)BT * 164 * sizeof(float));

    // ---------------- Conv stack (3 conv iters + groupnorm iter) -----------
    for (int i = 0; i < 3; ++i) {
        const float* src; int lds, coffC, cinC, coffF, cinF;
        if (i == 0) { src = IN(0); lds = 265; coffC = 0; cinC = 8;   coffF = 8;   cinF = 257; }
        else        { src = (i == 1) ? cfA : cfB; lds = 768; coffC = 0; cinC = 512; coffF = 512; cinF = 256; }
        float* dst = (i == 0) ? cfA : ((i == 1) ? cfB : cfA);
        int KcPad = pad32(cinC * 5);
        int KfPad = pad32(cinF * 5);

        long tc = (long)BT * KcPad;
        im2col5_kernel<<<ew_grid(tc), 256, 0, stream>>>(src, lds, coffC, cinC, KcPad, Aim, T, tc);
        long twc = (long)512 * KcPad;
        cvt_convw_kernel<<<ew_grid(twc), 256, 0, stream>>>(IN(3 + i), cinC, KcPad, Wbuf, twc);
        launch_gemm(Aim, KcPad, Wbuf, KcPad, dst + 0, 768, BT, 512, KcPad, IN(6 + i), 512, 1, stream);

        long tf = (long)BT * KfPad;
        im2col5_kernel<<<ew_grid(tf), 256, 0, stream>>>(src, lds, coffF, cinF, KfPad, Aim, T, tf);
        long twf = (long)256 * KfPad;
        cvt_convw_kernel<<<ew_grid(twf), 256, 0, stream>>>(IN(11 + i), cinF, KfPad, Wbuf, twf);
        launch_gemm(Aim, KfPad, Wbuf, KfPad, dst + 512, 768, BT, 256, KfPad, IN(14 + i), 256, 1, stream);
        // interp_lnr approximated as identity pass-through (MAX_PAD == T)
    }
    // i == 3: relu(group_norm(...)) on cfA, both halves, then identity interp
    groupnorm_relu_kernel<<<dim3(B, 32), 256, 0, stream>>>(cfA, 768, 0,   512, T, IN(9),  IN(10));
    groupnorm_relu_kernel<<<dim3(B, 16), 256, 0, stream>>>(cfA, 768, 512, 256, T, IN(17), IN(18));

    // ---------------- Encoder bi-LSTM c (H=8, in=512) ----------------------
    {
        long ta = (long)BT * 512;
        cvt_pad_kernel<<<ew_grid(ta), 256, 0, stream>>>(cfA, 768, 0, 512, 512, Aim, ta);
        long tw = (long)32 * 512;
        cvt_w_kernel<<<ew_grid(tw), 256, 0, stream>>>(IN(23), 32, 512, 512, Wbuf, tw);
        launch_gemm(Aim, 512, Wbuf, 512, xsF, 32, BT, 32, 512, IN(25), 32, 0, stream);
        cvt_w_kernel<<<ew_grid(tw), 256, 0, stream>>>(IN(26), 32, 512, 512, Wbuf, tw);
        launch_gemm(Aim, 512, Wbuf, 512, xsB, 32, BT, 32, 512, IN(28), 32, 0, stream);
        small_lstm_kernel<<<dim3(B, 2), 128, 0, stream>>>(xsF, xsB, 32, IN(24), IN(27), co, 16, 8, T);
    }
    // ---------------- Encoder bi-LSTM f (H=32, in=256) ---------------------
    {
        long ta = (long)BT * 256;
        cvt_pad_kernel<<<ew_grid(ta), 256, 0, stream>>>(cfA, 768, 512, 256, 256, Aim, ta);
        long tw = (long)128 * 256;
        cvt_w_kernel<<<ew_grid(tw), 256, 0, stream>>>(IN(29), 128, 256, 256, Wbuf, tw);
        launch_gemm(Aim, 256, Wbuf, 256, xsF, 128, BT, 128, 256, IN(31), 128, 0, stream);
        cvt_w_kernel<<<ew_grid(tw), 256, 0, stream>>>(IN(32), 128, 256, 256, Wbuf, tw);
        launch_gemm(Aim, 256, Wbuf, 256, xsB, 128, BT, 128, 256, IN(34), 128, 0, stream);
        small_lstm_kernel<<<dim3(B, 2), 128, 0, stream>>>(xsF, xsB, 128, IN(30), IN(33), fo, 64, 32, T);
    }
    // ---------------- r path: conv_r + GN + bi-LSTM (H=1, in=128) ----------
    {
        long tc = (long)BT * 64;
        im2col5_kernel<<<ew_grid(tc), 256, 0, stream>>>(IN(1), 8, 0, 8, 64, Aim, T, tc);
        long tw = (long)128 * 64;
        cvt_convw_kernel<<<ew_grid(tw), 256, 0, stream>>>(IN(19), 8, 64, Wbuf, tw);
        launch_gemm(Aim, 64, Wbuf, 64, rbuf, 128, BT, 128, 64, IN(20), 128, 1, stream);
        groupnorm_relu_kernel<<<dim3(B, 8), 256, 0, stream>>>(rbuf, 128, 0, 128, T, IN(21), IN(22));

        long ta = (long)BT * 128;
        cvt_pad_kernel<<<ew_grid(ta), 256, 0, stream>>>(rbuf, 128, 0, 128, 128, Aim, ta);
        long tw2 = (long)16 * 128;
        cvt_w_kernel<<<ew_grid(tw2), 256, 0, stream>>>(IN(35), 4, 128, 128, Wbuf, tw2);
        launch_gemm(Aim, 128, Wbuf, 128, xsF, 16, BT, 16, 128, IN(37), 4, 0, stream);
        cvt_w_kernel<<<ew_grid(tw2), 256, 0, stream>>>(IN(38), 4, 128, 128, Wbuf, tw2);
        launch_gemm(Aim, 128, Wbuf, 128, xsB, 16, BT, 16, 128, IN(40), 4, 0, stream);
        small_lstm_kernel<<<dim3(B, 2), 128, 0, stream>>>(xsF, xsB, 16, IN(36), IN(39), ro, 2, 1, T);
    }
    // ---------------- Build decoder input (B,T,164) ------------------------
    {
        long te = (long)BT * 164;
        build_enc_kernel<<<ew_grid(te), 256, 0, stream>>>(co, ro, fo, IN(2), encF, T, te);
        long th = (long)BT * 192;
        cvt_pad_kernel<<<ew_grid(th), 256, 0, stream>>>(encF, 164, 0, 164, 192, encH, th);
    }
    // ---------------- Decoder: 3 bidirectional LSTM layers (H=512) ---------
    {
        const int wih[3][2]  = {{41, 44}, {47, 50}, {53, 56}};
        const int uhh[3][2]  = {{42, 45}, {48, 51}, {54, 57}};
        const int bih[3][2]  = {{43, 46}, {49, 52}, {55, 58}};
        const int KinReal[3] = {164, 1024, 1024};
        const int KinPad[3]  = {192, 1024, 1024};
        const _Float16* Ain = encH;
        int ldaIn = 192;
        for (int l = 0; l < 3; ++l) {
            long tw = (long)2048 * KinPad[l];
            cvt_w_kernel<<<ew_grid(tw), 256, 0, stream>>>(IN(wih[l][0]), 2048, KinReal[l], KinPad[l], Wbuf, tw);
            launch_gemm(Ain, ldaIn, Wbuf, KinPad[l], xsF, 2048, BT, 2048, KinPad[l], IN(bih[l][0]), 2048, 0, stream);
            cvt_w_kernel<<<ew_grid(tw), 256, 0, stream>>>(IN(wih[l][1]), 2048, KinReal[l], KinPad[l], Wbuf, tw);
            launch_gemm(Ain, ldaIn, Wbuf, KinPad[l], xsB, 2048, BT, 2048, KinPad[l], IN(bih[l][1]), 2048, 0, stream);

            long tu = (long)2048 * 512;
            cvt_w_kernel<<<ew_grid(tu), 256, 0, stream>>>(IN(uhh[l][0]), 2048, 512, 512, whhF, tu);
            cvt_w_kernel<<<ew_grid(tu), 256, 0, stream>>>(IN(uhh[l][1]), 2048, 512, 512, whhB, tu);

            _Float16* Aout = (l == 1) ? actE1 : actE0;
            decoder_lstm_kernel<<<2, 1024, 0, stream>>>(xsF, xsB, whhF, whhB, Aout, T);
            Ain = Aout;
            ldaIn = 1024;
        }
        // final linear 1024 -> 80 into d_out (B,T,80) f32
        long tw = (long)80 * 1024;
        cvt_w_kernel<<<ew_grid(tw), 256, 0, stream>>>(IN(59), 80, 1024, 1024, Wbuf, tw);
        launch_gemm(Ain, 1024, Wbuf, 1024, (float*)d_out, 80, BT, 80, 1024, IN(60), 80, 0, stream);
    }
}
